// AAttn_13477607375836
// MI455X (gfx1250) — matrix operand order repaired and checked by reading, not compiled, not run on hardware
//
#include <hip/hip_runtime.h>

// ---------------------------------------------------------------------------
// MI455X (gfx1250) area-attention block.
//  x[8,256,64,64] -> qk(1x1 conv) / v(1x1 conv) -> depthwise5x5(v) = pp
//  flash attention over 32 areas x 8 heads x 1024 tokens x d=32
//  out = proj1x1(o + pp)
// All GEMM-shaped math uses v_wmma_f32_16x16x32_bf16 (bf16 in, f32 accum).
// K/V tile staging in attention uses the Tensor Data Mover with DOUBLE
// BUFFERING: tensor_load_to_lds for tile it+1 is issued before computing
// tile it, synchronized with s_wait_tensorcnt + workgroup barrier.
// Workspace: qk bf16 (33.5MB) | v bf16 (16.8MB) | o+pp f32 (33.5MB) ~ 84MB,
// which fits in the 192MB L2 -> inter-kernel traffic stays on-chip.
// ---------------------------------------------------------------------------

typedef float        v8f   __attribute__((ext_vector_type(8)));
typedef __bf16       v16bf __attribute__((ext_vector_type(16)));
typedef unsigned int v4u   __attribute__((ext_vector_type(4)));
typedef unsigned int v8u   __attribute__((ext_vector_type(8)));

__device__ __forceinline__ unsigned short f2bfu(float f) {
    __bf16 h = (__bf16)f;
    return __builtin_bit_cast(unsigned short, h);
}
__device__ __forceinline__ __bf16 bfu(unsigned short u) {
    return __builtin_bit_cast(__bf16, u);
}
__device__ __forceinline__ float bfu2f(unsigned short u) {
    return (float)__builtin_bit_cast(__bf16, u);
}
__device__ __forceinline__ v8f z8() {
    v8f r;
#pragma unroll
    for (int i = 0; i < 8; ++i) r[i] = 0.0f;
    return r;
}

// Issue one TDM tile load: 2D tile described by D# group1 'g1', destination
// LDS byte offset 'lds', source global address 'gsrc'.
__device__ __forceinline__ void tdm_load_tile(const unsigned short* gsrc,
                                              unsigned lds, const v8u& g1) {
    const unsigned long long ga = (unsigned long long)gsrc;
    v4u g0;
    g0[0] = 1u;                                   // count=1, user mode
    g0[1] = lds;                                  // lds_addr
    g0[2] = (unsigned)ga;                         // global_addr[31:0]
    g0[3] = (unsigned)(ga >> 32) | (2u << 30);    // global_addr[56:32] | type=2
    asm volatile("tensor_load_to_lds %0, %1" :: "s"(g0), "s"(g1) : "memory");
}

// ---------------------------------------------------------------------------
// Kernel 1 & 4: 1x1 conv as WMMA GEMM.
//   out[ch, n] = affine( sum_k W[ch,k] * X[k,n] ),  per batch (blockIdx.z).
// Block tile 128(M) x 128(N), K step 32. 8 waves: 2(M) x 4(N), each wave owns
// 64x32 = 4x2 WMMA 16x16 f32 accumulators.
// X tile stored TRANSPOSED in LDS ([n][k], 96B row stride) so B-fragment
// reads vectorize to ds_load_b128 like the A side.
// mode 0: rows [0,Msplit) -> bf16 out0 (512ch qk), rows >= Msplit -> bf16 out1
// mode 1: f32 outf (256 ch)
// ---------------------------------------------------------------------------
__global__ __launch_bounds__(256)
void conv1x1_wmma(const float* __restrict__ W0, const float* __restrict__ W1,
                  int Msplit,
                  const float* __restrict__ X,
                  const float* __restrict__ s0, const float* __restrict__ b0,
                  const float* __restrict__ s1, const float* __restrict__ b1,
                  unsigned short* __restrict__ out0,
                  unsigned short* __restrict__ out1,
                  float* __restrict__ outf,
                  int K, int mode)
{
    constexpr int N = 4096;
    constexpr int BSTR = 48;                     // padded k-stride (96B, 16B-aligned slices)
    __shared__ unsigned short As[128 * 32];      // [m][k] bf16 weight tile (8KB)
    __shared__ unsigned short Bs[128 * BSTR];    // [n][k] bf16 X tile, transposed (12KB)

    const int tid  = threadIdx.x;
    const int lane = tid & 31;
    const int wave = tid >> 5;
    const int wm   = wave >> 2;     // 0..1
    const int wn   = wave & 3;      // 0..3
    const bool hi  = lane >= 16;
    const int l16  = lane & 15;

    const int bz = blockIdx.z;
    const int m0 = blockIdx.y * 128;
    const int n0 = blockIdx.x * 128;

    const float* Xb = X + (size_t)bz * K * N;

    v8f acc[4][2];
#pragma unroll
    for (int mi = 0; mi < 4; ++mi)
#pragma unroll
        for (int ni = 0; ni < 2; ++ni) acc[mi][ni] = z8();

    // cooperative staging map: 16 contiguous global elements per thread
    const int ar = tid >> 1;            // 0..127 (weight row within tile)
    const int ak = (tid & 1) * 16;      // 0 / 16
    const int xr = tid >> 3;            // 0..31  (k row of X tile)
    const int xn = (tid & 7) * 16;      // n offset

    for (int k0 = 0; k0 < K; k0 += 32) {
        __syncthreads();
        {
            const int ch = m0 + ar;
            const float* Wp = (ch < Msplit) ? (W0 + (size_t)ch * K)
                                            : (W1 + (size_t)(ch - Msplit) * K);
#pragma unroll
            for (int i = 0; i < 16; ++i)
                As[ar * 32 + ak + i] = f2bfu(Wp[k0 + ak + i]);

            const float* Xp = Xb + (size_t)(k0 + xr) * N + n0 + xn;
#pragma unroll
            for (int i = 0; i < 16; ++i)                 // transpose on store
                Bs[(xn + i) * BSTR + xr] = f2bfu(Xp[i]);

            if (k0 + 32 < K)   // prefetch next X tile -> global_prefetch_b8
                __builtin_prefetch(Xb + (size_t)(k0 + 32 + xr) * N + n0 + xn, 0, 3);
        }
        __syncthreads();

        // B fragments: 32x16 bf16; lane&15 = column, lane half splits K.
        // Transposed LDS layout -> 16 consecutive ushorts -> 2x ds_load_b128.
        v16bf bfrag[2];
#pragma unroll
        for (int ni = 0; ni < 2; ++ni) {
            const int col = wn * 32 + ni * 16 + l16;
#pragma unroll
            for (int e = 0; e < 16; ++e) {
                const int kk = e + (hi ? 16 : 0);
                bfrag[ni][e] = bfu(Bs[col * BSTR + kk]);
            }
        }
        // A fragments: 16x32 bf16, contiguous K per lane half (matches B)
#pragma unroll
        for (int mi = 0; mi < 4; ++mi) {
            v16bf afrag;
            const int m = wm * 64 + mi * 16 + l16;
#pragma unroll
            for (int e = 0; e < 16; ++e) {
                const int kk = e + (hi ? 16 : 0);
                afrag[e] = bfu(As[m * 32 + kk]);
            }
#pragma unroll
            for (int ni = 0; ni < 2; ++ni)
                acc[mi][ni] = __builtin_amdgcn_wmma_f32_16x16x32_bf16(
                    false, afrag, false, bfrag[ni], (short)0, acc[mi][ni],
                    false, false);
        }
    }

    // Epilogue: C/D layout row = r + 8*half, col = lane&15
#pragma unroll
    for (int mi = 0; mi < 4; ++mi) {
#pragma unroll
        for (int ni = 0; ni < 2; ++ni) {
#pragma unroll
            for (int r = 0; r < 8; ++r) {
                const int ch = m0 + wm * 64 + mi * 16 + r + (hi ? 8 : 0);
                const int n  = n0 + wn * 32 + ni * 16 + l16;
                float v = acc[mi][ni][r];
                if (mode == 0) {
                    if (ch < Msplit) {
                        v = v * s0[ch] + b0[ch];
                        out0[((size_t)bz * 512 + ch) * N + n] = f2bfu(v);
                    } else {
                        const int c2 = ch - Msplit;
                        v = v * s1[c2] + b1[c2];
                        out1[((size_t)bz * 256 + c2) * N + n] = f2bfu(v);
                    }
                } else {
                    v = v * s0[ch] + b0[ch];
                    outf[((size_t)bz * 256 + ch) * N + n] = v;
                }
            }
        }
    }
}

// ---------------------------------------------------------------------------
// Kernel 2: depthwise 5x5 'SAME' conv on bf16 v, writes pp (f32) into o+pp buf.
// ---------------------------------------------------------------------------
__global__ __launch_bounds__(256)
void dwconv5x5(const unsigned short* __restrict__ vbf,
               const float* __restrict__ wpe,
               const float* __restrict__ spe, const float* __restrict__ bpe,
               float* __restrict__ opp)
{
    const int idx = blockIdx.x * 256 + threadIdx.x;   // < 8*256*4096
    const int n = idx & 4095;
    const int c = (idx >> 12) & 255;
    const int b = idx >> 20;
    const int y = n >> 6, x = n & 63;

    const unsigned short* vp = vbf + ((size_t)(b * 256 + c) << 12);
    const float* wp = wpe + c * 25;

    float acc = 0.0f;
#pragma unroll
    for (int dy = -2; dy <= 2; ++dy) {
        const int yy = y + dy;
        if (yy < 0 || yy > 63) continue;
#pragma unroll
        for (int dx = -2; dx <= 2; ++dx) {
            const int xx = x + dx;
            if (xx < 0 || xx > 63) continue;
            acc += bfu2f(vp[yy * 64 + xx]) * wp[(dy + 2) * 5 + (dx + 2)];
        }
    }
    opp[idx] = acc * spe[c] + bpe[c];
}

// ---------------------------------------------------------------------------
// Kernel 3: flash attention per (batch, area, head). Na=1024, d=32.
// Block = 8 waves x 16 q-rows = 128 q-rows. Key loop: 32 keys/chunk.
// K/V tiles (32 d-rows x 32 keys, row stride 4096 elems) are fetched by the
// Tensor Data Mover into DOUBLE-BUFFERED LDS tiles: tile it+1 is issued
// before computing tile it so the DMA overlaps the WMMAs + softmax.
// Per chunk/wave: 2 WMMA (S = Q K^T), online softmax with shuffle row
// reductions, P staged through per-wave LDS (C->A layout), 2 WMMA (O += P V).
// ---------------------------------------------------------------------------
__global__ __launch_bounds__(256)
void area_attn(const unsigned short* __restrict__ qkbf,
               const unsigned short* __restrict__ vbf,
               float* __restrict__ opp)
{
    constexpr int N = 4096;
    __shared__ unsigned short Kt[2][32 * 32];    // ping-pong [d][key]
    __shared__ unsigned short Vt[2][32 * 32];    // ping-pong [d][key]
    __shared__ unsigned short Pt[8][16 * 32];    // per-wave P tiles [row][key]

    const int tid  = threadIdx.x;
    const int lane = tid & 31;
    const int wave = tid >> 5;
    const bool hi  = lane >= 16;
    const int l16  = lane & 15;

    const int combo = blockIdx.y;        // 256 = 8 batches * 4 areas * 8 heads
    const int b = combo >> 5;
    const int a = (combo >> 3) & 3;
    const int h = combo & 7;

    const unsigned short* qb = qkbf + ((size_t)(b * 512 + h * 32)) * N + a * 1024;
    const unsigned short* kb = qkbf + ((size_t)(b * 512 + 256 + h * 32)) * N + a * 1024;
    const unsigned short* vb = vbf  + ((size_t)(b * 256 + h * 32)) * N + a * 1024;
    float*                ob = opp  + ((size_t)(b * 256 + h * 32)) * N + a * 1024;

    const int q0 = blockIdx.x * 128 + wave * 16;

    // Q A-fragment (constant across the key loop); d stride = 4096 channels
    v16bf qa;
    {
        const int m = q0 + l16;
#pragma unroll
        for (int e = 0; e < 16; ++e) {
            const int d = e + (hi ? 16 : 0);
            qa[e] = bfu(qb[(size_t)d * N + m]);
        }
    }

    v8f o0 = z8(), o1 = z8();
    float mrow[8], lrow[8];
#pragma unroll
    for (int r = 0; r < 8; ++r) { mrow[r] = -1.0e30f; lrow[r] = 0.0f; }

    const float scale = 0.17677669529663688f;   // 32^-0.5

    // Tensor DMA descriptor group 1 (shared by K and V tiles, loop-invariant):
    //   data_size=2B, tensor_dim0=1024 (keys in area), tensor_dim1=32 (d rows),
    //   tile 32x32, tensor_dim0_stride=4096 elements, no padding/iterate/mask.
    v8u g1;
    g1[0] = 0x00010000u;     // workgroup_mask=0 | data_size=1 (2 bytes)
    g1[1] = 1024u << 16;     // tensor_dim0[15:0] at bits 63:48
    g1[2] = 32u << 16;       // tensor_dim0[31:16]=0 | tensor_dim1[15:0]=32
    g1[3] = 32u << 16;       // tensor_dim1[31:16]=0 | tile_dim0=32
    g1[4] = 32u;             // tile_dim1=32 | tile_dim2=0
    g1[5] = 4096u;           // tensor_dim0_stride[31:0]
    g1[6] = 0u;              // stride0[47:32]=0 | tensor_dim1_stride[15:0]=0
    g1[7] = 0u;

    const unsigned ldsK[2] = { (unsigned)(unsigned long long)&Kt[0][0],
                               (unsigned)(unsigned long long)&Kt[1][0] };
    const unsigned ldsV[2] = { (unsigned)(unsigned long long)&Vt[0][0],
                               (unsigned)(unsigned long long)&Vt[1][0] };

    // prologue: start DMA of tile 0
    if (wave == 0) {
        tdm_load_tile(kb, ldsK[0], g1);
        tdm_load_tile(vb, ldsV[0], g1);
    }

    for (int it = 0; it < 32; ++it) {
        const int buf = it & 1;
        // wave0: tile 'it' DMA complete (issued last iteration / prologue).
        // No-op for other waves (TENSORcnt==0).
        __builtin_amdgcn_s_wait_tensorcnt(0);
        __syncthreads();   // publish tile 'it'; all waves done with buf 'it-1'

        // kick off DMA of tile it+1 into the other buffer; overlaps compute
        if (wave == 0 && it + 1 < 32) {
            const int key1 = (it + 1) * 32;
            tdm_load_tile(kb + key1, ldsK[buf ^ 1], g1);
            tdm_load_tile(vb + key1, ldsV[buf ^ 1], g1);
        }

        const unsigned short* Ktc = &Kt[buf][0];
        const unsigned short* Vtc = &Vt[buf][0];

        // S = Q K^T  (two 16x16 tiles covering 32 keys)
        v8f s0 = z8(), s1 = z8();
        {
            v16bf bk0, bk1;
#pragma unroll
            for (int e = 0; e < 16; ++e) {
                const int d = e + (hi ? 16 : 0);
                bk0[e] = bfu(Ktc[d * 32 + l16]);
                bk1[e] = bfu(Ktc[d * 32 + 16 + l16]);
            }
            s0 = __builtin_amdgcn_wmma_f32_16x16x32_bf16(false, qa, false, bk0,
                                                         (short)0, s0, false, false);
            s1 = __builtin_amdgcn_wmma_f32_16x16x32_bf16(false, qa, false, bk1,
                                                         (short)0, s1, false, false);
        }

        // online softmax; row reductions over the 16-lane half via xor shuffles
        float alpha[8];
#pragma unroll
        for (int r = 0; r < 8; ++r) {
            float v0 = s0[r] * scale, v1 = s1[r] * scale;
            float mx = fmaxf(v0, v1);
            mx = fmaxf(mx, __shfl_xor(mx, 1, 32));
            mx = fmaxf(mx, __shfl_xor(mx, 2, 32));
            mx = fmaxf(mx, __shfl_xor(mx, 4, 32));
            mx = fmaxf(mx, __shfl_xor(mx, 8, 32));
            const float mn = fmaxf(mrow[r], mx);
            alpha[r] = __expf(mrow[r] - mn);
            mrow[r] = mn;
            const float p0 = __expf(v0 - mn);
            const float p1 = __expf(v1 - mn);
            s0[r] = p0; s1[r] = p1;
            float rs = p0 + p1;
            rs += __shfl_xor(rs, 1, 32);
            rs += __shfl_xor(rs, 2, 32);
            rs += __shfl_xor(rs, 4, 32);
            rs += __shfl_xor(rs, 8, 32);
            lrow[r] = lrow[r] * alpha[r] + rs;
            o0[r] *= alpha[r];
            o1[r] *= alpha[r];
        }

        // P (C layout) -> LDS (row-major) so we can refetch in A layout
#pragma unroll
        for (int r = 0; r < 8; ++r) {
            const int row = r + (hi ? 8 : 0);
            Pt[wave][row * 32 + l16]      = f2bfu(s0[r]);
            Pt[wave][row * 32 + 16 + l16] = f2bfu(s1[r]);
        }
        __syncthreads();

        // O += P * V   (A = P 16x32 keys, B = V 32keys x 16d, two d-halves)
        v16bf pa, bv0, bv1;
#pragma unroll
        for (int e = 0; e < 16; ++e) {
            const int kk = e + (hi ? 16 : 0);
            pa[e] = bfu(Pt[wave][l16 * 32 + kk]);
            const int key = e + (hi ? 16 : 0);
            bv0[e] = bfu(Vtc[l16 * 32 + key]);
            bv1[e] = bfu(Vtc[(16 + l16) * 32 + key]);
        }
        o0 = __builtin_amdgcn_wmma_f32_16x16x32_bf16(false, pa, false, bv0,
                                                     (short)0, o0, false, false);
        o1 = __builtin_amdgcn_wmma_f32_16x16x32_bf16(false, pa, false, bv1,
                                                     (short)0, o1, false, false);
    }

    // normalize and accumulate onto pp (each element has exactly one owner)
#pragma unroll
    for (int r = 0; r < 8; ++r) {
        const float inv = 1.0f / lrow[r];
        const int qrow = q0 + r + (hi ? 8 : 0);
        float* p0 = ob + (size_t)l16 * N + qrow;
        float* p1 = ob + (size_t)(16 + l16) * N + qrow;
        *p0 += o0[r] * inv;
        *p1 += o1[r] * inv;
    }
}

// ---------------------------------------------------------------------------
extern "C" void kernel_launch(void* const* d_in, const int* in_sizes, int n_in,
                              void* d_out, int out_size, void* d_ws, size_t ws_size,
                              hipStream_t stream)
{
    (void)in_sizes; (void)n_in; (void)out_size; (void)ws_size;

    const float* x      = (const float*)d_in[0];
    const float* w_qk   = (const float*)d_in[1];
    const float* s_qk   = (const float*)d_in[2];
    const float* b_qk   = (const float*)d_in[3];
    const float* w_v    = (const float*)d_in[4];
    const float* s_v    = (const float*)d_in[5];
    const float* b_v    = (const float*)d_in[6];
    const float* w_pe   = (const float*)d_in[7];
    const float* s_pe   = (const float*)d_in[8];
    const float* b_pe   = (const float*)d_in[9];
    const float* w_proj = (const float*)d_in[10];
    const float* s_proj = (const float*)d_in[11];
    const float* b_proj = (const float*)d_in[12];
    float* out = (float*)d_out;

    char* ws = (char*)d_ws;
    unsigned short* qkb = (unsigned short*)ws;                              // 8*512*4096 bf16
    unsigned short* vb  = (unsigned short*)(ws + (size_t)33554432);         // 8*256*4096 bf16
    float*          opp = (float*)(ws + (size_t)33554432 + 16777216);       // 8*256*4096 f32

    const dim3 blk(256, 1, 1);

    // 1) fused qk|v 1x1 conv: M=768 (512 qk + 256 v), K=256, N=4096, 8 batches
    conv1x1_wmma<<<dim3(32, 6, 8), blk, 0, stream>>>(
        w_qk, w_v, 512, x, s_qk, b_qk, s_v, b_v, qkb, vb, nullptr, 256, 0);

    // 2) depthwise 5x5 on v -> pp (seeds the o+pp buffer)
    dwconv5x5<<<dim3(32768, 1, 1), blk, 0, stream>>>(vb, w_pe, s_pe, b_pe, opp);

    // 3) flash attention: 8 q-blocks x (8 batch * 4 area * 8 head) combos
    area_attn<<<dim3(8, 256, 1), blk, 0, stream>>>(qkb, vb, opp);

    // 4) projection 1x1 conv on (o + pp) -> f32 output
    conv1x1_wmma<<<dim3(32, 2, 8), blk, 0, stream>>>(
        w_proj, w_proj, 1 << 30, opp, s_proj, b_proj, s_proj, b_proj,
        nullptr, nullptr, out, 256, 1);
}
